// MultiViewSpaTracker_90563680403794
// MI455X (gfx1250) — matrix operand order repaired and checked by reading, not compiled, not run on hardware
//
#include <hip/hip_runtime.h>
#include <math.h>

// ---------------------------------------------------------------------------
// MultiViewSpaTracker forward for MI455X (gfx1250, wave32, WMMA).
// Dominant cost = correlation GEMMs (M=128,N=16384,K=128, batched over S=8,
// 3 views x 4 pyramid levels x 2 iters ~= 34 GFLOP) -> v_wmma_f32_16x16x32_f16
// with f32 accumulate. Transformer GEMMs (1024 rows) reuse the same kernel.
// GEMM is software-pipelined with double-buffered LDS: global loads for tile
// k+1 are issued before the WMMAs on tile k, so load latency hides under the
// matrix ops; one barrier per K-step. Fragments are fetched from LDS as
// 128-bit DS loads (contiguous per-lane layout, B staged transposed).
// ---------------------------------------------------------------------------

typedef _Float16 f16;
typedef __attribute__((ext_vector_type(4)))  _Float16 v4h;
typedef __attribute__((ext_vector_type(8)))  _Float16 v8h;
typedef __attribute__((ext_vector_type(16))) _Float16 v16h;
typedef __attribute__((ext_vector_type(8)))  float    v8f;

#define S_     8
#define N_     128
#define LAT_   128
#define HID_   384
#define NH_    8
#define NL_    6
#define T_     8
#define DIN_   1169
#define DINP_  1172   // DIN padded to multiple of 4 for vector staging
#define DOUT_  387
#define DH_    48
#define ITERS_ 2

#define BM 64
#define BN 128
#define BK 32

static inline int cdiv(long a, int b) { return (int)((a + b - 1) / b); }

// ---------------------------------------------------------------------------
// Batched GEMM: C[bz] = alpha * A[bz] x B[bz] + bias + Cadd
// 256 threads = 8 waves (2x4); block tile 64x128; wave tile 32x32 (2x2 WMMA
// tiles -> 4 wmma per K-step with 2x fragment reuse).
// ---------------------------------------------------------------------------
__global__ __launch_bounds__(256) void gemm_wmma(
    const float* __restrict__ A, const float* __restrict__ B,
    float* __restrict__ C, const float* __restrict__ bias,
    const float* __restrict__ Cadd,
    int M, int N, int K, int lda, int ldb, int ldc,
    long sA, long sB, long sC, float alpha)
{
  __shared__ __align__(16) f16 As[2][BM][BK];    // 8 KB (double-buffered)
  __shared__ __align__(16) f16 Bst[2][BN][BK];   // 16 KB, transposed B

  const int bz = blockIdx.z;
  A += (long)bz * sA;
  B += (long)bz * sB;
  C += (long)bz * sC;
  if (Cadd) Cadd += (long)bz * sC;

  const int m0   = blockIdx.y * BM;
  const int n0   = blockIdx.x * BN;
  const int tid  = threadIdx.x;
  const int lane = tid & 31;
  const int wave = tid >> 5;
  const int wm   = wave >> 2;   // 0..1 -> 32-row strip
  const int wn   = wave & 3;    // 0..3 -> 32-col strip

  v8f acc[2][2] = {{{}, {}}, {{}, {}}};

  const int arow  = lane & 15;          // A fragment row within 16-row tile
  const int kgrp  = (lane >> 4) * 16;   // A: upper lane-half holds K+16 group
  const int bcol  = lane & 15;          // B fragment column
  const int bkgrp = (lane >> 4) * 16;   // B: upper lane-half holds K+16 group

  // Per-thread staging slice coordinates (float4 granularity).
  const int aR0 = (tid * 4) >> 5,          aC0 = (tid * 4) & 31;
  const int aR1 = (1024 + tid * 4) >> 5,   aC1 = (1024 + tid * 4) & 31;
  const int bR0 = (tid * 4) >> 7,          bC0 = (tid * 4) & 127;
  const int bR1 = (1024 + tid * 4) >> 7,   bC1 = (1024 + tid * 4) & 127;

  // Fragment compute + WMMA on buffer `buf` (shared by both paths).
  auto mma_step = [&](int buf) {
    v16h a[2], b[2];
#pragma unroll
    for (int tm = 0; tm < 2; ++tm) {
      const f16* p = &As[buf][wm * 32 + tm * 16 + arow][0];
      v8h lo = *(const v8h*)(p + kgrp);
      v8h hi = *(const v8h*)(p + kgrp + 8);
      a[tm] = __builtin_shufflevector(lo, hi, 0, 1, 2, 3, 4, 5, 6, 7,
                                      8, 9, 10, 11, 12, 13, 14, 15);
    }
#pragma unroll
    for (int tn = 0; tn < 2; ++tn) {
      const f16* p = &Bst[buf][wn * 32 + tn * 16 + bcol][0];
      v8h lo = *(const v8h*)(p + bkgrp);
      v8h hi = *(const v8h*)(p + bkgrp + 8);
      b[tn] = __builtin_shufflevector(lo, hi, 0, 1, 2, 3, 4, 5, 6, 7,
                                      8, 9, 10, 11, 12, 13, 14, 15);
    }
#pragma unroll
    for (int tm = 0; tm < 2; ++tm)
#pragma unroll
      for (int tn = 0; tn < 2; ++tn)
        acc[tm][tn] = __builtin_amdgcn_wmma_f32_16x16x32_f16(
            false, a[tm], false, b[tn], (short)0, acc[tm][tn], false, false);
  };

  const bool fastMN = (m0 + BM <= M) && (n0 + BN <= N) &&
                      ((lda & 3) == 0) && ((ldb & 3) == 0);

  if (fastMN && ((K & (BK - 1)) == 0)) {
    // ---- software-pipelined path: double-buffered LDS ----
    float4 va0, va1, vb0, vb1;
    auto load_tile = [&](int k0) {   // batch all 4 global b128 loads
      va0 = *(const float4*)(A + (long)(m0 + aR0) * lda + k0 + aC0);
      va1 = *(const float4*)(A + (long)(m0 + aR1) * lda + k0 + aC1);
      vb0 = *(const float4*)(B + (long)(k0 + bR0) * ldb + n0 + bC0);
      vb1 = *(const float4*)(B + (long)(k0 + bR1) * ldb + n0 + bC1);
    };
    auto store_tile = [&](int buf) { // f32->f16 convert at LDS store
      v4h h0, h1;
      h0[0] = (f16)va0.x; h0[1] = (f16)va0.y; h0[2] = (f16)va0.z; h0[3] = (f16)va0.w;
      h1[0] = (f16)va1.x; h1[1] = (f16)va1.y; h1[2] = (f16)va1.z; h1[3] = (f16)va1.w;
      *(v4h*)&As[buf][aR0][aC0] = h0;          // ds_store_b64
      *(v4h*)&As[buf][aR1][aC1] = h1;
      Bst[buf][bC0 + 0][bR0] = (f16)vb0.x;
      Bst[buf][bC0 + 1][bR0] = (f16)vb0.y;
      Bst[buf][bC0 + 2][bR0] = (f16)vb0.z;
      Bst[buf][bC0 + 3][bR0] = (f16)vb0.w;
      Bst[buf][bC1 + 0][bR1] = (f16)vb1.x;
      Bst[buf][bC1 + 1][bR1] = (f16)vb1.y;
      Bst[buf][bC1 + 2][bR1] = (f16)vb1.z;
      Bst[buf][bC1 + 3][bR1] = (f16)vb1.w;
    };

    const int nk = K / BK;
    load_tile(0);
    store_tile(0);
    __syncthreads();
    for (int kb = 0; kb < nk; ++kb) {
      const int cur = kb & 1;
      if (kb + 1 < nk) {
        load_tile((kb + 1) * BK);    // issued before WMMAs; latency hidden
        if (kb + 2 < nk) {           // speculative L2 prefetch 2 tiles ahead
          __builtin_prefetch(A + (long)(m0 + aR0) * lda + (kb + 2) * BK + aC0, 0, 1);
          __builtin_prefetch(B + (long)((kb + 2) * BK + bR0) * ldb + n0 + bC0, 0, 1);
        }
      }
      mma_step(cur);
      if (kb + 1 < nk) store_tile(cur ^ 1);
      __syncthreads();
    }
  } else {
    // ---- generic path: bounds-checked scalar staging, single buffer ----
    for (int k0 = 0; k0 < K; k0 += BK) {
      for (int i = tid; i < BM * BK; i += 256) {
        int r = i / BK, c = i % BK;
        int gr = m0 + r, gc = k0 + c;
        As[0][r][c] = (f16)((gr < M && gc < K) ? A[(long)gr * lda + gc] : 0.0f);
      }
      for (int i = tid; i < BK * BN; i += 256) {
        int r = i / BN, c = i % BN;              // r = k, c = col
        int gr = k0 + r, gc = n0 + c;
        Bst[0][c][r] = (f16)((gr < K && gc < N) ? B[(long)gr * ldb + gc] : 0.0f);
      }
      __syncthreads();
      mma_step(0);
      __syncthreads();
    }
  }

  // Store: C/D layout -> VGPR r holds M = r (lanes 0-15) / r+8 (lanes 16-31).
#pragma unroll
  for (int tm = 0; tm < 2; ++tm) {
#pragma unroll
    for (int tn = 0; tn < 2; ++tn) {
      const int rbase = m0 + wm * 32 + tm * 16 + ((lane < 16) ? 0 : 8);
      const int col   = n0 + wn * 32 + tn * 16 + (lane & 15);
      if (col >= N) continue;
#pragma unroll
      for (int r = 0; r < 8; ++r) {
        int row = rbase + r;
        if (row >= M) continue;
        float v = alpha * acc[tm][tn][r];
        if (bias) v += bias[col];
        if (Cadd) v += Cadd[(long)row * ldc + col];
        C[(long)row * ldc + col] = v;
      }
    }
  }
}

static void gemm(hipStream_t st, const float* A, const float* B, float* C,
                 const float* bias, const float* Cadd,
                 int M, int N, int K, int lda, int ldb, int ldc,
                 long sA, long sB, long sC, int batch, float alpha)
{
  dim3 g(cdiv(N, BN), cdiv(M, BM), batch);
  gemm_wmma<<<g, 256, 0, st>>>(A, B, C, bias, Cadd, M, N, K, lda, ldb, ldc,
                               sA, sB, sC, alpha);
}

// ---------------------------------------------------------------------------
// 2x2 average pooling over (planes, H, W) f32.
// ---------------------------------------------------------------------------
__global__ void avgpool2_kernel(const float* __restrict__ in,
                                float* __restrict__ out, int planes, int H, int W)
{
  int Ho = H >> 1, Wo = W >> 1;
  long total = (long)planes * Ho * Wo;
  long i = (long)blockIdx.x * 256 + threadIdx.x;
  if (i >= total) return;
  int wo = (int)(i % Wo);
  long t = i / Wo;
  int ho = (int)(t % Ho);
  int p  = (int)(t / Ho);
  const float* r0 = in + ((long)p * H + 2 * ho) * W + 2 * wo;
  out[i] = 0.25f * (r0[0] + r0[1] + r0[W] + r0[W + 1]);
}

// feat_init (S,N,LAT,3) -> ffeat_k (S,N,LAT)
__global__ void extract_feat_kernel(const float* __restrict__ fi,
                                    float* __restrict__ f, int k)
{
  long total = (long)S_ * N_ * LAT_;
  long i = (long)blockIdx.x * 256 + threadIdx.x;
  if (i >= total) return;
  f[i] = fi[i * 3 + k];
}

// ---------------------------------------------------------------------------
// Bilinear sampling of corr (S, N, H*W) at coords/2^l + delta (7x7 window),
// writing into fcorr (N, S, 588) at channel chan0 + p.
// ---------------------------------------------------------------------------
__global__ void corr_sample_kernel(const float* __restrict__ corr,
                                   const float* __restrict__ coords,
                                   float* __restrict__ fc,
                                   int H, int W, int level, int cx, int cy,
                                   int chan0)
{
  long total = (long)S_ * N_ * 49;
  long i = (long)blockIdx.x * 256 + threadIdx.x;
  if (i >= total) return;
  int p = (int)(i % 49);
  long sn = i / 49;
  int n = (int)(sn % N_);
  int s = (int)(sn / N_);

  float inv = 1.0f / (float)(1 << level);
  const float* cd = coords + ((long)s * N_ + n) * 3;
  float x = cd[cx] * inv + (float)(p / 7 - 3);
  float y = cd[cy] * inv + (float)(p % 7 - 3);

  const float* cf = corr + ((long)s * N_ + n) * H * W;
  int x0 = (int)floorf(x), y0 = (int)floorf(y);
  float wx = x - (float)x0, wy = y - (float)y0;

  auto g = [&](int xi, int yi) -> float {
    if (xi < 0 || xi >= W || yi < 0 || yi >= H) return 0.0f;
    return cf[(long)yi * W + xi];
  };
  float v = g(x0, y0)         * (1.f - wx) * (1.f - wy)
          + g(x0 + 1, y0)     * wx         * (1.f - wy)
          + g(x0, y0 + 1)     * (1.f - wx) * wy
          + g(x0 + 1, y0 + 1) * wx         * wy;
  fc[((long)n * S_ + s) * 588 + chan0 + p] = v;
}

// ---------------------------------------------------------------------------
// Assemble transformer input x (N, S, DINP) (last DINP-DIN cols zero):
//   [flow_embed(195) | fcorrs(588) | ffeats(384) | tmv(2)] + pos + time.
// ---------------------------------------------------------------------------
__global__ void assemble_x_kernel(const float* __restrict__ coords,
                                  const float* __restrict__ coords_init,
                                  const float* __restrict__ fc,
                                  const float* __restrict__ f0,
                                  const float* __restrict__ f1,
                                  const float* __restrict__ f2,
                                  const float* __restrict__ track_mask,
                                  const float* __restrict__ vis_init,
                                  float* __restrict__ x)
{
  long total = (long)N_ * S_ * DINP_;
  long i = (long)blockIdx.x * 256 + threadIdx.x;
  if (i >= total) return;
  int c = (int)(i % DINP_);
  long ns = i / DINP_;
  int s = (int)(ns % S_);
  int n = (int)(ns / S_);

  if (c >= DIN_) {            // zero padding columns
    x[((long)n * S_ + s) * DINP_ + c] = 0.0f;
    return;
  }

  float val;
  if (c < 195) {
    if (c < 192) {
      int a = c / 64, rem = c % 64, j = rem >> 1;
      float fa = coords[((long)s * N_ + n) * 3 + a] - coords[(long)n * 3 + a];
      float ang = fa * ((float)j * 31.25f);      // (2j) * (1000/64)
      val = (rem & 1) ? cosf(ang) : sinf(ang);
    } else {
      int a = c - 192;
      val = coords[((long)s * N_ + n) * 3 + a] - coords[(long)n * 3 + a];
    }
  } else if (c < 783) {
    val = fc[((long)n * S_ + s) * 588 + (c - 195)];
  } else if (c < 1167) {
    int k = (c - 783) / LAT_, cc = (c - 783) % LAT_;
    const float* fk = (k == 0) ? f0 : (k == 1) ? f1 : f2;
    val = fk[((long)s * N_ + n) * LAT_ + cc];
  } else if (c == 1167) {
    val = track_mask[(long)s * N_ + n];
  } else {
    val = vis_init[(long)s * N_ + n];
  }

  // positional embed from coords_init frame 0 (3 x sincos1d(390))[:DIN]
  {
    int d3 = c / 390;
    int r  = c - d3 * 390;
    float gi = (coords_init[(long)n * 3 + d3] / 128.0f * 2.0f - 1.0f) * 128.0f;
    int r195 = (r < 195) ? r : r - 195;
    float omega = powf(10000.0f, -(float)r195 / 195.0f);
    val += (r < 195) ? sinf(gi * omega) : cosf(gi * omega);
  }
  // time embed sincos1d(1170, s)[:DIN]
  {
    int rc = (c < 585) ? c : c - 585;
    float omega = powf(10000.0f, -(float)rc / 585.0f);
    val += (c < 585) ? sinf((float)s * omega) : cosf((float)s * omega);
  }
  x[((long)n * S_ + s) * DINP_ + c] = val;
}

// ---------------------------------------------------------------------------
// Layernorm / groupnorm over C channels; strided input, contiguous output.
// ---------------------------------------------------------------------------
__global__ void layernorm_kernel(const float* __restrict__ in, int ldin,
                                 const float* __restrict__ g,
                                 const float* __restrict__ b,
                                 float* __restrict__ out, int C)
{
  int row = blockIdx.x;
  const float* xr = in + (long)row * ldin;
  __shared__ float red[128];
  float s = 0.0f;
  for (int c = threadIdx.x; c < C; c += 128) s += xr[c];
  red[threadIdx.x] = s;
  __syncthreads();
  for (int o = 64; o > 0; o >>= 1) {
    if ((int)threadIdx.x < o) red[threadIdx.x] += red[threadIdx.x + o];
    __syncthreads();
  }
  float mean = red[0] / (float)C;
  __syncthreads();
  float v = 0.0f;
  for (int c = threadIdx.x; c < C; c += 128) {
    float d = xr[c] - mean;
    v += d * d;
  }
  red[threadIdx.x] = v;
  __syncthreads();
  for (int o = 64; o > 0; o >>= 1) {
    if ((int)threadIdx.x < o) red[threadIdx.x] += red[threadIdx.x + o];
    __syncthreads();
  }
  float inv = rsqrtf(red[0] / (float)C + 1e-5f);
  for (int c = threadIdx.x; c < C; c += 128)
    out[(long)row * C + c] = (xr[c] - mean) * inv * g[c] + b[c];
}

// ---------------------------------------------------------------------------
// Attention: qkv (G*T, 3*HID) -> o (G*T, HID). One block per (group, head),
// one thread per query token; online (flash-style) softmax over T <= 128.
// ---------------------------------------------------------------------------
__global__ void attention_kernel(const float* __restrict__ qkv,
                                 float* __restrict__ o, int G, int T)
{
  int gh = blockIdx.x;
  int g = gh / NH_, h = gh % NH_;
  int t = threadIdx.x;
  if (t >= T) return;
  const float scale = rsqrtf((float)DH_);
  const float* q = qkv + ((long)(g * T + t) * 3) * HID_ + h * DH_;

  float m = -1e30f, l = 0.0f;
  float out[DH_];
#pragma unroll
  for (int e = 0; e < DH_; ++e) out[e] = 0.0f;

  for (int s = 0; s < T; ++s) {
    const float* kr = qkv + ((long)(g * T + s) * 3 + 1) * HID_ + h * DH_;
    const float* vr = qkv + ((long)(g * T + s) * 3 + 2) * HID_ + h * DH_;
    float d = 0.0f;
    for (int e = 0; e < DH_; ++e) d += q[e] * kr[e];
    d *= scale;
    float mn = fmaxf(m, d);
    float cf = __expf(m - mn);
    float p  = __expf(d - mn);
    for (int e = 0; e < DH_; ++e) out[e] = out[e] * cf + p * vr[e];
    l = l * cf + p;
    m = mn;
  }
  float inv = 1.0f / l;
  float* orow = o + (long)(g * T + t) * HID_ + h * DH_;
  for (int e = 0; e < DH_; ++e) orow[e] = out[e] * inv;
}

// exact GELU, in place
__global__ void gelu_kernel(float* __restrict__ x, long n)
{
  long i = (long)blockIdx.x * 256 + threadIdx.x;
  if (i >= n) return;
  float v = x[i];
  x[i] = 0.5f * v * (1.0f + erff(v * 0.70710678118f));
}

// (P,T,C) -> (T,P,C)
__global__ void transpose_pt_kernel(const float* __restrict__ in,
                                    float* __restrict__ out, int P, int T, int C)
{
  long total = (long)P * T * C;
  long i = (long)blockIdx.x * 256 + threadIdx.x;
  if (i >= total) return;
  int c = (int)(i % C);
  long pt = i / C;
  int t = (int)(pt % T);
  int p = (int)(pt / T);
  out[((long)t * P + p) * C + c] = in[((long)p * T + t) * C + c];
}

// cbias[c] = b_in[c] + mean_rows(sproj)[c]
__global__ void bias_mean_kernel(const float* __restrict__ b_in,
                                 const float* __restrict__ sproj,
                                 float* __restrict__ out)
{
  int c = blockIdx.x * 256 + threadIdx.x;
  if (c >= HID_) return;
  float s = 0.0f;
  for (int r = 0; r < 100; ++r) s += sproj[(long)r * HID_ + c];
  out[c] = b_in[c] + s * 0.01f;
}

__global__ void support_update_kernel(float* __restrict__ support,
                                      const float* __restrict__ sproj)
{
  long total = 100L * HID_;
  long i = (long)blockIdx.x * 256 + threadIdx.x;
  if (i >= total) return;
  support[i] += sproj[i] * 0.01f;
}

// coords (S,N,3) += delta[(n*T+s), 0..2]
__global__ void update_coords_kernel(float* __restrict__ coords,
                                     const float* __restrict__ delta)
{
  long total = (long)S_ * N_ * 3;
  long i = (long)blockIdx.x * 256 + threadIdx.x;
  if (i >= total) return;
  int c = (int)(i % 3);
  long sn = i / 3;
  int n = (int)(sn % N_);
  int s = (int)(sn / N_);
  coords[i] += delta[((long)n * T_ + s) * DOUT_ + c];
}

// ffeat_k (S,N,LAT) += upd[(n*T+s), LAT]
__global__ void update_ffeat_kernel(float* __restrict__ f,
                                    const float* __restrict__ upd)
{
  long total = (long)S_ * N_ * LAT_;
  long i = (long)blockIdx.x * 256 + threadIdx.x;
  if (i >= total) return;
  int c = (int)(i % LAT_);
  long sn = i / LAT_;
  int n = (int)(sn % N_);
  int s = (int)(sn / N_);
  f[i] += upd[((long)n * T_ + s) * LAT_ + c];
}

// visfeat (S*N, 384) = concat(f0,f1,f2)
__global__ void concat_feats_kernel(const float* __restrict__ f0,
                                    const float* __restrict__ f1,
                                    const float* __restrict__ f2,
                                    float* __restrict__ vf)
{
  long total = (long)S_ * N_ * HID_;
  long i = (long)blockIdx.x * 256 + threadIdx.x;
  if (i >= total) return;
  int c = (int)(i % HID_);
  long r = i / HID_;
  int k = c / LAT_, cc = c % LAT_;
  const float* fk = (k == 0) ? f0 : (k == 1) ? f1 : f2;
  vf[i] = fk[r * LAT_ + cc];
}

// out (S,N,4) = [coords, vis]
__global__ void final_out_kernel(const float* __restrict__ coords,
                                 const float* __restrict__ vis,
                                 float* __restrict__ out)
{
  long total = (long)S_ * N_ * 4;
  long i = (long)blockIdx.x * 256 + threadIdx.x;
  if (i >= total) return;
  int j = (int)(i % 4);
  long sn = i / 4;
  out[i] = (j < 3) ? coords[sn * 3 + j] : vis[sn];
}

// ---------------------------------------------------------------------------
// Host-side orchestration.
// ---------------------------------------------------------------------------
extern "C" void kernel_launch(void* const* d_in, const int* in_sizes, int n_in,
                              void* d_out, int out_size, void* d_ws, size_t ws_size,
                              hipStream_t stream)
{
  (void)in_sizes; (void)n_in; (void)out_size; (void)ws_size;

  const float* fmap[3]     = {(const float*)d_in[0], (const float*)d_in[1],
                              (const float*)d_in[2]};
  const float* coords_init = (const float*)d_in[3];
  const float* vis_init    = (const float*)d_in[4];
  const float* track_mask  = (const float*)d_in[5];
  const float* feat_init   = (const float*)d_in[6];
  const float* support_in  = (const float*)d_in[7];
  const float* W_in  = (const float*)d_in[9];
  const float* b_in  = (const float*)d_in[10];
  const float* lnp   = (const float*)d_in[11];
  const float* Wqkv  = (const float*)d_in[12];
  const float* bqkv  = (const float*)d_in[13];
  const float* Wo    = (const float*)d_in[14];
  const float* bo    = (const float*)d_in[15];
  const float* W1    = (const float*)d_in[16];
  const float* b1    = (const float*)d_in[17];
  const float* W2    = (const float*)d_in[18];
  const float* b2    = (const float*)d_in[19];
  const float* W_out = (const float*)d_in[20];
  const float* b_out = (const float*)d_in[21];
  const float* Ws    = (const float*)d_in[22];
  const float* bs    = (const float*)d_in[23];
  const float* gn_g  = (const float*)d_in[24];
  const float* gn_b  = (const float*)d_in[25];
  const float* Wu    = (const float*)d_in[26];
  const float* bu    = (const float*)d_in[27];
  const float* Wvis  = (const float*)d_in[28];
  const float* bvis  = (const float*)d_in[29];

  // workspace carve-out
  char* wsp = (char*)d_ws;
  size_t off = 0;
  auto alloc = [&](long nfloats) -> float* {
    float* p = (float*)(wsp + off);
    off += (((size_t)nfloats * sizeof(float)) + 255) & ~(size_t)255;
    return p;
  };

  const int dims[4] = {128, 64, 32, 16};
  const float* pyr[3][4];
  for (int v = 0; v < 3; ++v) {
    pyr[v][0] = fmap[v];
    for (int l = 1; l < 4; ++l)
      pyr[v][l] = alloc((long)S_ * LAT_ * dims[l] * dims[l]);
  }
  float* corrbuf = alloc((long)S_ * N_ * 128 * 128);
  float* fcorr   = alloc((long)N_ * S_ * 588);
  float* ffeat[3];
  for (int k = 0; k < 3; ++k) ffeat[k] = alloc((long)S_ * N_ * LAT_);
  float* coords  = alloc((long)S_ * N_ * 3);
  float* support = alloc(100L * HID_);
  float* sproj   = alloc(100L * HID_);
  float* cbias   = alloc(HID_);
  float* xbuf    = alloc(1024L * DINP_);
  float* h       = alloc(1024L * HID_);
  float* htr     = alloc(1024L * HID_);
  float* lnbuf   = alloc(1024L * HID_);
  float* qkvbuf  = alloc(1024L * 3 * HID_);
  float* obuf    = alloc(1024L * HID_);
  float* ffnbuf  = alloc(1024L * 4 * HID_);
  float* delta   = alloc(1024L * DOUT_);
  float* gnbuf   = alloc(1024L * LAT_);
  float* updbuf  = alloc(1024L * LAT_);
  float* visfeat = alloc(1024L * HID_);
  float* visout  = alloc(1024L);

  // init mutable state (never touch d_in in place)
  hipMemcpyAsync(coords, coords_init, (size_t)S_ * N_ * 3 * sizeof(float),
                 hipMemcpyDeviceToDevice, stream);
  hipMemcpyAsync(support, support_in, (size_t)100 * HID_ * sizeof(float),
                 hipMemcpyDeviceToDevice, stream);
  for (int k = 0; k < 3; ++k)
    extract_feat_kernel<<<cdiv((long)S_ * N_ * LAT_, 256), 256, 0, stream>>>(
        feat_init, ffeat[k], k);

  // build pyramids (f32)
  for (int v = 0; v < 3; ++v)
    for (int l = 1; l < 4; ++l)
      avgpool2_kernel<<<cdiv((long)S_ * LAT_ * dims[l] * dims[l], 256), 256, 0,
                        stream>>>(pyr[v][l - 1], (float*)pyr[v][l],
                                  S_ * LAT_, dims[l - 1], dims[l - 1]);

  const float corr_scale = 0.08838834764831845f;  // 1/sqrt(128)

  for (int it = 0; it < ITERS_; ++it) {
    // --- correlation pyramids + bilinear sampling (dominant GEMMs) ---
    for (int v = 0; v < 3; ++v) {
      int cx = (v == 0) ? 0 : (v == 1) ? 1 : 0;
      int cy = (v == 0) ? 1 : (v == 1) ? 2 : 2;
      for (int l = 0; l < 4; ++l) {
        int d = dims[l], hw = d * d;
        gemm(stream, ffeat[v], pyr[v][l], corrbuf, nullptr, nullptr,
             N_, hw, LAT_, LAT_, hw, hw,
             (long)N_ * LAT_, (long)LAT_ * hw, (long)N_ * hw, S_, corr_scale);
        corr_sample_kernel<<<cdiv((long)S_ * N_ * 49, 256), 256, 0, stream>>>(
            corrbuf, coords, fcorr, d, d, l, cx, cy, v * 196 + l * 49);
      }
    }

    // --- assemble transformer input ---
    assemble_x_kernel<<<cdiv((long)N_ * S_ * DINP_, 256), 256, 0, stream>>>(
        coords, coords_init, fcorr, ffeat[0], ffeat[1], ffeat[2],
        track_mask, vis_init, xbuf);

    // support projection + combined bias
    gemm(stream, support, Ws, sproj, bs, nullptr, 100, HID_, HID_,
         HID_, HID_, HID_, 0, 0, 0, 1, 1.0f);
    bias_mean_kernel<<<cdiv(HID_, 256), 256, 0, stream>>>(b_in, sproj, cbias);

    // input projection (A row stride padded to DINP_ for vector staging)
    gemm(stream, xbuf, W_in, h, cbias, nullptr, 1024, HID_, DIN_,
         DINP_, HID_, HID_, 0, 0, 0, 1, 1.0f);

    // --- 6 alternating (time / track) transformer layers ---
    for (int i = 0; i < NL_; ++i) {
      float* cur;
      int G, Ttok;
      if ((i & 1) == 0) { cur = h; G = N_; Ttok = T_; }
      else {
        transpose_pt_kernel<<<cdiv(1024L * HID_, 256), 256, 0, stream>>>(
            h, htr, N_, T_, HID_);
        cur = htr; G = T_; Ttok = N_;
      }
      const float* g1  = lnp + ((long)i * 4 + 0) * HID_;
      const float* bb1 = lnp + ((long)i * 4 + 1) * HID_;
      const float* g2  = lnp + ((long)i * 4 + 2) * HID_;
      const float* bb2 = lnp + ((long)i * 4 + 3) * HID_;

      layernorm_kernel<<<1024, 128, 0, stream>>>(cur, HID_, g1, bb1, lnbuf, HID_);
      gemm(stream, lnbuf, Wqkv + (long)i * HID_ * 3 * HID_, qkvbuf,
           bqkv + (long)i * 3 * HID_, nullptr, 1024, 3 * HID_, HID_,
           HID_, 3 * HID_, 3 * HID_, 0, 0, 0, 1, 1.0f);
      attention_kernel<<<G * NH_, 128, 0, stream>>>(qkvbuf, obuf, G, Ttok);
      gemm(stream, obuf, Wo + (long)i * HID_ * HID_, cur,
           bo + (long)i * HID_, cur, 1024, HID_, HID_,
           HID_, HID_, HID_, 0, 0, 0, 1, 1.0f);

      layernorm_kernel<<<1024, 128, 0, stream>>>(cur, HID_, g2, bb2, lnbuf, HID_);
      gemm(stream, lnbuf, W1 + (long)i * HID_ * 4 * HID_, ffnbuf,
           b1 + (long)i * 4 * HID_, nullptr, 1024, 4 * HID_, HID_,
           HID_, 4 * HID_, 4 * HID_, 0, 0, 0, 1, 1.0f);
      gelu_kernel<<<cdiv(1024L * 4 * HID_, 256), 256, 0, stream>>>(
          ffnbuf, 1024L * 4 * HID_);
      gemm(stream, ffnbuf, W2 + (long)i * 4 * HID_ * HID_, cur,
           b2 + (long)i * HID_, cur, 1024, HID_, 4 * HID_,
           4 * HID_, HID_, HID_, 0, 0, 0, 1, 1.0f);

      if ((i & 1) == 1)
        transpose_pt_kernel<<<cdiv(1024L * HID_, 256), 256, 0, stream>>>(
            htr, h, T_, N_, HID_);
    }

    // output projection + state updates
    gemm(stream, h, W_out, delta, b_out, nullptr, 1024, DOUT_, HID_,
         HID_, DOUT_, DOUT_, 0, 0, 0, 1, 1.0f);
    support_update_kernel<<<cdiv(100L * HID_, 256), 256, 0, stream>>>(support, sproj);
    update_coords_kernel<<<cdiv((long)S_ * N_ * 3, 256), 256, 0, stream>>>(coords, delta);

    for (int k = 0; k < 3; ++k) {
      layernorm_kernel<<<1024, 128, 0, stream>>>(
          delta + 3 + k * LAT_, DOUT_, gn_g + (long)k * LAT_,
          gn_b + (long)k * LAT_, gnbuf, LAT_);
      gemm(stream, gnbuf, Wu + (long)k * LAT_ * LAT_, updbuf,
           bu + (long)k * LAT_, nullptr, 1024, LAT_, LAT_,
           LAT_, LAT_, LAT_, 0, 0, 0, 1, 1.0f);
      gelu_kernel<<<cdiv(1024L * LAT_, 256), 256, 0, stream>>>(updbuf, 1024L * LAT_);
      update_ffeat_kernel<<<cdiv((long)S_ * N_ * LAT_, 256), 256, 0, stream>>>(
          ffeat[k], updbuf);
    }
  }

  // visibility head + final packing
  concat_feats_kernel<<<cdiv(1024L * HID_, 256), 256, 0, stream>>>(
      ffeat[0], ffeat[1], ffeat[2], visfeat);
  gemm(stream, visfeat, Wvis, visout, bvis, nullptr, 1024, 1, HID_,
       HID_, 1, 1, 0, 0, 0, 1, 1.0f);
  final_out_kernel<<<cdiv(4096L, 256), 256, 0, stream>>>(
      coords, visout, (float*)d_out);
}
